// MultiHeadSelfAttention_44753559225020
// MI455X (gfx1250) — matrix operand order repaired and checked by reading, not compiled, not run on hardware
//
#include <hip/hip_runtime.h>

typedef __attribute__((ext_vector_type(16))) _Float16 v16h;
typedef __attribute__((ext_vector_type(8)))  float    v8f;
typedef __attribute__((ext_vector_type(2)))  _Float16 h2v;

#define WMMA_F16(a, b, c) \
  __builtin_amdgcn_wmma_f32_16x16x32_f16(false, (a), false, (b), (short)0, (c), false, false)

// ---------------- constants ----------------
#define BATCH 8
#define HH 56
#define WW 56
#define CH 384
#define HEADS 6
#define DH 64
#define NQ 3136      // 56*56
#define NKV 784      // 28*28
#define NKVP 800     // padded to multiple of 32
#define QT_PER (NQ/16)   // 196

// ---------------- fragment loaders (wave32 WMMA layouts) ----------------
// A 16x32 f16: lane m=lane&15, hh=lane>>4; VGPR v: K=16*hh+2v,2v+1
__device__ __forceinline__ v16h load_a_frag(const _Float16* __restrict__ arow, int k0, int hh) {
  v16h a;
#pragma unroll
  for (int v = 0; v < 8; ++v) {
    int kb = k0 + 16 * hh + 2 * v;
    h2v p = *(const h2v*)(arow + kb);
    a[2 * v] = p[0];
    a[2 * v + 1] = p[1];
  }
  return a;
}

// B 32x16 f16 from BT[N][K] row-major: lane n=lane&15, hh=lane>>4; VGPR v: K=16*hh+2v,2v+1
// caller passes brow = BT + n*ldb + 16*hh (+ col/base offsets)
__device__ __forceinline__ v16h load_b_frag(const _Float16* __restrict__ brow, int k0) {
  v16h b;
#pragma unroll
  for (int v = 0; v < 8; ++v) {
    h2v p = *(const h2v*)(brow + k0 + 2 * v);
    b[2 * v] = p[0];
    b[2 * v + 1] = p[1];
  }
  return b;
}

// ---------------- weight transpose fp32[K][N] -> f16 BT[N][K] ----------------
__global__ void transpose_w_kernel(const float* __restrict__ w, _Float16* __restrict__ bt,
                                   int K, int N) {
  int t = blockIdx.x * blockDim.x + threadIdx.x;
  if (t >= K * N) return;
  int n = t % N, k = t / N;
  bt[n * K + k] = (_Float16)w[t];
}

// ---------------- depthwise 3x3 + BN -> f16 activations ----------------
__global__ void dw_bn_kernel(const float* __restrict__ x, const float* __restrict__ wdw,
                             const float* __restrict__ gamma, const float* __restrict__ beta,
                             const float* __restrict__ mean, const float* __restrict__ var,
                             _Float16* __restrict__ out, int Ho, int Wo, int stride, int pad) {
  int t = blockIdx.x * blockDim.x + threadIdx.x;
  int total = BATCH * Ho * Wo * CH;
  if (t >= total) return;
  int c = t % CH;
  int wo = (t / CH) % Wo;
  int ho = (t / (CH * Wo)) % Ho;
  int b = t / (CH * Wo * Ho);
  float acc = 0.f;
#pragma unroll
  for (int kh = 0; kh < 3; ++kh) {
    int ih = ho * stride + kh - pad;
    if (ih < 0 || ih >= HH) continue;
#pragma unroll
    for (int kw = 0; kw < 3; ++kw) {
      int iw = wo * stride + kw - pad;
      if (iw < 0 || iw >= WW) continue;
      acc += x[((b * HH + ih) * WW + iw) * CH + c] * wdw[(kh * 3 + kw) * CH + c];
    }
  }
  float inv = rsqrtf(var[c] + 1e-5f);
  float yv = (acc - mean[c]) * (gamma[c] * inv) + beta[c];
  out[t] = (_Float16)yv;
}

// ---------------- core 32x64 macro-tile: 8 independent WMMAs per k-step ----------------
// acc[mi][ni] covers rows row0+mi*16, cols col0+ni*16
__device__ __forceinline__ void wmma_tile_32x64(const _Float16* __restrict__ A, int lda,
                                                const _Float16* __restrict__ BT, int ldb,
                                                int row0, int col0, int K,
                                                int mr, int hh, v8f acc[2][4]) {
  const _Float16* arow0 = A + (size_t)(row0 + mr) * lda;
  const _Float16* arow1 = A + (size_t)(row0 + 16 + mr) * lda;
  const _Float16* brow = BT + (size_t)(col0 + mr) * ldb + 16 * hh;
  for (int k0 = 0; k0 < K; k0 += 32) {
    if (k0 + 64 < K) {  // WGP-scope prefetch of next-next-step operands
      __builtin_prefetch(arow0 + k0 + 64, 0, 3);
      __builtin_prefetch(arow1 + k0 + 64, 0, 3);
      __builtin_prefetch(brow + k0 + 64, 0, 3);
    }
    v16h a0 = load_a_frag(arow0, k0, hh);
    v16h a1 = load_a_frag(arow1, k0, hh);
    v16h b0 = load_b_frag(brow, k0);
    v16h b1 = load_b_frag(brow + (size_t)16 * ldb, k0);
    v16h b2 = load_b_frag(brow + (size_t)32 * ldb, k0);
    v16h b3 = load_b_frag(brow + (size_t)48 * ldb, k0);
    acc[0][0] = WMMA_F16(a0, b0, acc[0][0]);
    acc[0][1] = WMMA_F16(a0, b1, acc[0][1]);
    acc[0][2] = WMMA_F16(a0, b2, acc[0][2]);
    acc[0][3] = WMMA_F16(a0, b3, acc[0][3]);
    acc[1][0] = WMMA_F16(a1, b0, acc[1][0]);
    acc[1][1] = WMMA_F16(a1, b1, acc[1][1]);
    acc[1][2] = WMMA_F16(a1, b2, acc[1][2]);
    acc[1][3] = WMMA_F16(a1, b3, acc[1][3]);
  }
}

// ---------------- Q pointwise GEMM: yq[25088,384] x BTq -> Qh[bh][nq][64] (scaled) ------
__global__ void gemm_q_kernel(const _Float16* __restrict__ A, const _Float16* __restrict__ BT,
                              _Float16* __restrict__ Qh) {
  int wave = blockIdx.x * (blockDim.x >> 5) + (threadIdx.x >> 5);
  const int Nb = CH / 64;            // 6
  const int Mb = (BATCH * NQ) / 32;  // 784
  if (wave >= Mb * Nb) return;
  int rt = wave / Nb, ct = wave % Nb;
  int lane = threadIdx.x & 31, mr = lane & 15, hh = lane >> 4;
  v8f acc[2][4] = {};
  wmma_tile_32x64(A, CH, BT, CH, rt * 32, ct * 64, CH, mr, hh, acc);
  // 3136 % 32 == 0: a 32-row macro-tile never straddles an image
  int b = (rt * 32) / NQ;
  int nq0 = (rt * 32) % NQ;
#pragma unroll
  for (int mi = 0; mi < 2; ++mi)
#pragma unroll
    for (int ni = 0; ni < 4; ++ni) {
      int cc = ct * 64 + ni * 16 + mr;
      int head = cc >> 6, d = cc & 63;
      _Float16* qb = Qh + ((size_t)(b * HEADS + head) * NQ + nq0 + mi * 16 + 8 * hh) * DH + d;
#pragma unroll
      for (int i = 0; i < 8; ++i) qb[(size_t)i * DH] = (_Float16)(acc[mi][ni][i] * 0.125f);
    }
}

// ---------------- KV pointwise GEMM: ykv[6272,384] x BTkv -> Kh / Vt ----------------
__global__ void gemm_kv_kernel(const _Float16* __restrict__ A, const _Float16* __restrict__ BT,
                               _Float16* __restrict__ Kh, _Float16* __restrict__ Vt) {
  int wave = blockIdx.x * (blockDim.x >> 5) + (threadIdx.x >> 5);
  const int Nb = (2 * CH) / 64;       // 12
  const int Mb = (BATCH * NKV) / 32;  // 196
  if (wave >= Mb * Nb) return;
  int rt = wave / Nb, ct = wave % Nb;
  int lane = threadIdx.x & 31, mr = lane & 15, hh = lane >> 4;
  v8f acc[2][4] = {};
  wmma_tile_32x64(A, CH, BT, CH, rt * 32, ct * 64, CH, mr, hh, acc);
#pragma unroll
  for (int mi = 0; mi < 2; ++mi)
#pragma unroll
    for (int ni = 0; ni < 4; ++ni) {
      int cc = ct * 64 + ni * 16 + mr;
#pragma unroll
      for (int i = 0; i < 8; ++i) {
        int r = rt * 32 + mi * 16 + i + 8 * hh;  // 784 % 32 != 0: keep per-row split
        int b = r / NKV, nkv = r % NKV;
        if (cc < CH) {
          int head = cc >> 6, d = cc & 63;
          Kh[((size_t)(b * HEADS + head) * NKV + nkv) * DH + d] = (_Float16)acc[mi][ni][i];
        } else {
          int c2 = cc - CH;
          int head = c2 >> 6, d = c2 & 63;
          Vt[((size_t)(b * HEADS + head) * DH + d) * NKVP + nkv] = (_Float16)acc[mi][ni][i];
        }
      }
    }
}

// ---------------- zero the kv padding columns of Vt ----------------
__global__ void vt_pad_kernel(_Float16* __restrict__ Vt) {
  int t = blockIdx.x * blockDim.x + threadIdx.x;
  if (t >= BATCH * HEADS * DH * (NKVP - NKV)) return;
  int n = NKV + (t & 15);
  int rest = t >> 4;  // bh*64 + d
  Vt[(size_t)rest * NKVP + n] = (_Float16)0.f;
}

// ---------------- attention: one wave per 16 query rows, scores block in LDS --------
__global__ void attn_kernel(const _Float16* __restrict__ Qh, const _Float16* __restrict__ Kh,
                            const _Float16* __restrict__ Vt, _Float16* __restrict__ Oasm) {
  extern __shared__ float smem[];
  int wslot = threadIdx.x >> 5;
  float* S = smem + (size_t)wslot * 16 * NKVP;
  int wave = blockIdx.x * (blockDim.x >> 5) + wslot;
  if (wave >= BATCH * HEADS * QT_PER) return;
  int bh = wave / QT_PER, qt = wave % QT_PER;
  int b = bh / HEADS, head = bh % HEADS;
  int lane = threadIdx.x & 31, mr = lane & 15, hh = lane >> 4;

  // Q fragments (16 rows x 64 K) held for the whole kernel
  const _Float16* qrow = Qh + ((size_t)bh * NQ + qt * 16 + mr) * DH;
  v16h qa0 = load_a_frag(qrow, 0, hh);
  v16h qa1 = load_a_frag(qrow, 32, hh);

  // S = (Q*scale) K^T   (K rows are B^T directly)
  const _Float16* Kbase = Kh + (size_t)bh * NKV * DH;
  for (int j = 0; j < NKV / 16; ++j) {
    const _Float16* brow = Kbase + (size_t)(j * 16 + mr) * DH + 16 * hh;
    v16h b0 = load_b_frag(brow, 0);
    v16h b1 = load_b_frag(brow, 32);
    v8f acc = {};
    acc = WMMA_F16(qa0, b0, acc);
    acc = WMMA_F16(qa1, b1, acc);
#pragma unroll
    for (int i = 0; i < 8; ++i) S[(i + 8 * hh) * NKVP + j * 16 + mr] = acc[i];
  }
  // zero padding columns 784..799 (P contribution = 0)
  for (int idx = lane; idx < 16 * (NKVP - NKV); idx += 32)
    S[(idx >> 4) * NKVP + NKV + (idx & 15)] = 0.f;
  asm volatile("s_wait_dscnt 0x0" ::: "memory");

  // row-wise softmax, wave-cooperative
  for (int r = 0; r < 16; ++r) {
    float* row = S + r * NKVP;
    float m = -3.4e38f;
    for (int c = lane; c < NKV; c += 32) m = fmaxf(m, row[c]);
#pragma unroll
    for (int o = 16; o > 0; o >>= 1) m = fmaxf(m, __shfl_xor(m, o, 32));
    float s = 0.f;
    for (int c = lane; c < NKV; c += 32) {
      float e = __expf(row[c] - m);
      row[c] = e;
      s += e;
    }
#pragma unroll
    for (int o = 16; o > 0; o >>= 1) s += __shfl_xor(s, o, 32);
    float inv = 1.f / s;
    for (int c = lane; c < NKV; c += 32) row[c] *= inv;
  }
  asm volatile("s_wait_dscnt 0x0" ::: "memory");

  // O = P V  : K over 800 in chunks of 32, N = 64 in 4 chunks of 16
  const _Float16* Vbase = Vt + (size_t)bh * DH * NKVP;
  const float* prow = S + mr * NKVP;
  v8f oacc[4] = {};
  for (int j = 0; j < NKVP / 32; ++j) {
    int k0 = j * 32;
    v16h pa;
#pragma unroll
    for (int v = 0; v < 8; ++v) {
      int kb = k0 + 16 * hh + 2 * v;
      pa[2 * v] = (_Float16)prow[kb];
      pa[2 * v + 1] = (_Float16)prow[kb + 1];
    }
#pragma unroll
    for (int nc = 0; nc < 4; ++nc) {
      const _Float16* brow = Vbase + (size_t)(nc * 16 + mr) * NKVP + 16 * hh;
      v16h vb = load_b_frag(brow, k0);
      oacc[nc] = WMMA_F16(pa, vb, oacc[nc]);
    }
  }
  // write O in (b, xy, h*64+d) layout for the output projection
  _Float16* Obase = Oasm + ((size_t)b * NQ + qt * 16) * CH + head * DH;
#pragma unroll
  for (int nc = 0; nc < 4; ++nc)
#pragma unroll
    for (int i = 0; i < 8; ++i) {
      int r = i + 8 * hh;
      Obase[(size_t)r * CH + nc * 16 + mr] = (_Float16)oacc[nc][i];
    }
}

// ---------------- output projection + bias (fp32 out) ----------------
__global__ void gemm_out_kernel(const _Float16* __restrict__ A, const _Float16* __restrict__ BT,
                                const float* __restrict__ bias, float* __restrict__ out) {
  int wave = blockIdx.x * (blockDim.x >> 5) + (threadIdx.x >> 5);
  const int Nb = CH / 64;
  const int Mb = (BATCH * NQ) / 32;
  if (wave >= Mb * Nb) return;
  int rt = wave / Nb, ct = wave % Nb;
  int lane = threadIdx.x & 31, mr = lane & 15, hh = lane >> 4;
  v8f acc[2][4] = {};
  wmma_tile_32x64(A, CH, BT, CH, rt * 32, ct * 64, CH, mr, hh, acc);
#pragma unroll
  for (int mi = 0; mi < 2; ++mi)
#pragma unroll
    for (int ni = 0; ni < 4; ++ni) {
      float bv = bias[ct * 64 + ni * 16 + mr];
      float* ob = out + (size_t)(rt * 32 + mi * 16 + 8 * hh) * CH + ct * 64 + ni * 16 + mr;
#pragma unroll
      for (int i = 0; i < 8; ++i) ob[(size_t)i * CH] = acc[mi][ni][i] + bv;
    }
}

extern "C" void kernel_launch(void* const* d_in, const int* in_sizes, int n_in,
                              void* d_out, int out_size, void* d_ws, size_t ws_size,
                              hipStream_t stream) {
  (void)in_sizes; (void)n_in; (void)out_size; (void)ws_size;
  const float* x        = (const float*)d_in[0];
  const float* w_dw_q   = (const float*)d_in[1];
  const float* bn_q_g   = (const float*)d_in[2];
  const float* bn_q_b   = (const float*)d_in[3];
  const float* bn_q_m   = (const float*)d_in[4];
  const float* bn_q_v   = (const float*)d_in[5];
  const float* w_pw_q   = (const float*)d_in[6];
  const float* w_dw_kv  = (const float*)d_in[7];
  const float* bn_kv_g  = (const float*)d_in[8];
  const float* bn_kv_b  = (const float*)d_in[9];
  const float* bn_kv_m  = (const float*)d_in[10];
  const float* bn_kv_v  = (const float*)d_in[11];
  const float* w_pw_kv  = (const float*)d_in[12];
  const float* w_out    = (const float*)d_in[13];
  const float* b_out    = (const float*)d_in[14];
  float* out = (float*)d_out;

  char* ws = (char*)d_ws;
  size_t off = 0;
  auto alloc = [&](size_t bytes) {
    void* p = ws + off;
    off += (bytes + 255) & ~(size_t)255;
    return p;
  };
  _Float16* yq   = (_Float16*)alloc((size_t)BATCH * NQ * CH * 2);
  _Float16* ykv  = (_Float16*)alloc((size_t)BATCH * NKV * CH * 2);
  _Float16* Qh   = (_Float16*)alloc((size_t)BATCH * HEADS * NQ * DH * 2);
  _Float16* Kh   = (_Float16*)alloc((size_t)BATCH * HEADS * NKV * DH * 2);
  _Float16* Vt   = (_Float16*)alloc((size_t)BATCH * HEADS * DH * NKVP * 2);
  _Float16* BTq  = (_Float16*)alloc((size_t)CH * CH * 2);
  _Float16* BTkv = (_Float16*)alloc((size_t)CH * 2 * CH * 2);
  _Float16* BTo  = (_Float16*)alloc((size_t)CH * CH * 2);
  _Float16* Oasm = yq;  // yq is dead after gemm_q; reuse

  // weight transposes (fp32 -> f16 BT[N][K])
  transpose_w_kernel<<<(CH * CH + 255) / 256, 256, 0, stream>>>(w_pw_q, BTq, CH, CH);
  transpose_w_kernel<<<(CH * 2 * CH + 255) / 256, 256, 0, stream>>>(w_pw_kv, BTkv, CH, 2 * CH);
  transpose_w_kernel<<<(CH * CH + 255) / 256, 256, 0, stream>>>(w_out, BTo, CH, CH);

  // depthwise + BN
  int dwq_total = BATCH * HH * WW * CH;
  dw_bn_kernel<<<(dwq_total + 255) / 256, 256, 0, stream>>>(
      x, w_dw_q, bn_q_g, bn_q_b, bn_q_m, bn_q_v, yq, HH, WW, 1, 1);
  int dwkv_total = BATCH * (HH / 2) * (WW / 2) * CH;
  dw_bn_kernel<<<(dwkv_total + 255) / 256, 256, 0, stream>>>(
      x, w_dw_kv, bn_kv_g, bn_kv_b, bn_kv_m, bn_kv_v, ykv, HH / 2, WW / 2, 2, 0);

  // pointwise GEMMs (8 waves / block, one 32x64 macro-tile / wave)
  int qwaves = ((BATCH * NQ) / 32) * (CH / 64);        // 4704
  gemm_q_kernel<<<qwaves / 8, 256, 0, stream>>>(yq, BTq, Qh);

  vt_pad_kernel<<<(BATCH * HEADS * DH * (NKVP - NKV) + 255) / 256, 256, 0, stream>>>(Vt);
  int kvwaves = ((BATCH * NKV) / 32) * ((2 * CH) / 64);  // 2352
  gemm_kv_kernel<<<kvwaves / 8, 256, 0, stream>>>(ykv, BTkv, Kh, Vt);

  // attention: 2 waves / block, 16x800 fp32 score block per wave in LDS
  int awaves = BATCH * HEADS * QT_PER;
  size_t lds = (size_t)2 * 16 * NKVP * sizeof(float);  // 102400 B
  attn_kernel<<<awaves / 2, 64, lds, stream>>>(Qh, Kh, Vt, Oasm);

  // output projection + bias
  gemm_out_kernel<<<qwaves / 8, 256, 0, stream>>>(Oasm, BTo, b_out, out);
}
